// Decoder_10058813407462
// MI455X (gfx1250) — matrix operand order repaired and checked by reading, not compiled, not run on hardware
//
#include <hip/hip_runtime.h>
#include <hip/hip_bf16.h>
#include <math.h>

// ---------------------------------------------------------------------------
// Problem constants (from reference setup_inputs)
// ---------------------------------------------------------------------------
#define B   128
#define KV  15
#define KW  32
#define FV  2048
#define FW  300
#define FT  2348     // FV + FW
#define H   512
#define E   128
#define AH  512
#define V   10000
#define T   15
#define XD  2476     // E + FT
#define G4  2048     // 4*H

// padded K (multiple of 32) / padded N (multiple of 32) for bf16 buffers
#define FTp   2368   // ceil32(2348)
#define FWp   320    // ceil32(300)
#define XDp   2496   // ceil32(2476)
#define FTnp  2368   // Wgate rows padded
#define Vnp   10016  // Wout rows padded

typedef __attribute__((ext_vector_type(16))) __bf16          v16bf;
typedef __attribute__((ext_vector_type(16))) unsigned short  v16us;
typedef __attribute__((ext_vector_type(8)))  unsigned short  v8us;
typedef __attribute__((ext_vector_type(8)))  float           v8f;

__device__ __forceinline__ unsigned short f2bf(float f) {
  union { float f; unsigned u; } x; x.f = f;
  unsigned r = x.u + 0x7FFFu + ((x.u >> 16) & 1u);   // round-to-nearest-even
  return (unsigned short)(r >> 16);
}
__device__ __forceinline__ float sigm(float x) { return 1.0f / (1.0f + expf(-x)); }

// ---------------------------------------------------------------------------
// fp32 (R,K) -> bf16 (Rp,Kp), zero padded. Rows/K padded so the GEMM needs
// no load guards at all.
// ---------------------------------------------------------------------------
__global__ void k_f32_to_bf16_pad(const float* __restrict__ src,
                                  unsigned short* __restrict__ dst,
                                  int R, int K, int Kp, int total)
{
  int i = blockIdx.x * blockDim.x + threadIdx.x;
  if (i >= total) return;
  int r = i / Kp, k = i % Kp;
  dst[i] = (r < R && k < K) ? f2bf(src[(size_t)r * K + k]) : (unsigned short)0;
}

// ---------------------------------------------------------------------------
// WMMA GEMM on pre-converted bf16:  C[M,N] = (acc?C:0) + A[M,Kp]*W[N,Kp]^T + bias
// One wave computes a 16x32 C tile (2 WMMAs sharing one A fragment).
// M multiple of 16 (always true here); weight rows padded to >= tn+32;
// Kp multiple of 32. Only stores are guarded (ragged N=2348 / N=10000).
//
// Fragment layouts used here (wave32; contiguous k, same in A and B, not the ISA table order):
//  A (16x32 bf16): lane l -> row m = l&15; half = l>>4;
//     elements e: K = half*16 + e
//  B (32x16 bf16): lane l -> col n = l&15; elements e: K = half*16 + e
//  C/D (16x16 f32): lane l -> col n = l&15; VGPR r -> row m = r + (l>>4)*8
// ---------------------------------------------------------------------------
__global__ __launch_bounds__(32)
void gemm_bf16_wmma(const unsigned short* __restrict__ A,
                    const unsigned short* __restrict__ W,
                    const float* __restrict__ bias, float* __restrict__ C,
                    int M, int N, int Kp, int acc)
{
  const int lane = threadIdx.x;
  const int half = lane >> 4;
  const int idx  = lane & 15;
  const int tm = blockIdx.y * 16;
  const int tn = blockIdx.x * 32;
  const unsigned short* arow  = A + (size_t)(tm + idx) * Kp + half * 16;
  const unsigned short* wrow0 = W + (size_t)(tn + idx) * Kp + half * 16;
  const unsigned short* wrow1 = W + (size_t)(tn + 16 + idx) * Kp + half * 16;

  v8f c0 = {}, c1 = {};
  for (int k0 = 0; k0 < Kp; k0 += 32) {
    v8us alo  = *(const v8us*)(arow + k0);
    v8us ahi  = *(const v8us*)(arow + k0 + 8);
    v8us b0lo = *(const v8us*)(wrow0 + k0);
    v8us b0hi = *(const v8us*)(wrow0 + k0 + 8);
    v8us b1lo = *(const v8us*)(wrow1 + k0);
    v8us b1hi = *(const v8us*)(wrow1 + k0 + 8);
    // prefetch next k-step (global_prefetch_b8; harmless speculative)
    __builtin_prefetch(arow + k0 + 32, 0, 1);
    __builtin_prefetch(wrow0 + k0 + 32, 0, 1);
    __builtin_prefetch(wrow1 + k0 + 32, 0, 1);
    v16bf a  = __builtin_bit_cast(v16bf, __builtin_shufflevector(
        alo, ahi, 0,1,2,3,4,5,6,7,8,9,10,11,12,13,14,15));
    v16bf b0 = __builtin_bit_cast(v16bf, __builtin_shufflevector(
        b0lo, b0hi, 0,1,2,3,4,5,6,7,8,9,10,11,12,13,14,15));
    v16bf b1 = __builtin_bit_cast(v16bf, __builtin_shufflevector(
        b1lo, b1hi, 0,1,2,3,4,5,6,7,8,9,10,11,12,13,14,15));
    c0 = __builtin_amdgcn_wmma_f32_16x16x32_bf16(false, a, false, b0,
                                                 (short)0, c0, false, false);
    c1 = __builtin_amdgcn_wmma_f32_16x16x32_bf16(false, a, false, b1,
                                                 (short)0, c1, false, false);
  }
  const int n0 = tn + idx;
  const int n1 = tn + 16 + idx;
  const bool ok0 = n0 < N, ok1 = n1 < N;
  const float bb0 = (bias && ok0) ? bias[n0] : 0.0f;
  const float bb1 = (bias && ok1) ? bias[n1] : 0.0f;
  #pragma unroll
  for (int r = 0; r < 8; ++r) {
    int mm = tm + r + half * 8;
    float* crow = C + (size_t)mm * N;
    if (ok0) { float v = c0[r] + bb0; if (acc) v += crow[n0]; crow[n0] = v; }
    if (ok1) { float v = c1[r] + bb1; if (acc) v += crow[n1]; crow[n1] = v; }
  }
}

// ---------------------------------------------------------------------------
// Elementwise / small kernels
// ---------------------------------------------------------------------------
__global__ void k_init_prev(int* prev) { prev[threadIdx.x] = 0; }

__global__ void k_tanh_inplace(float* p, int n) {
  int i = blockIdx.x * blockDim.x + threadIdx.x;
  if (i < n) p[i] = tanhf(p[i]);
}

// pooled[b, :] = concat(mean_k features_v[b,k,:], mean_k features_w[b,k,:])
__global__ void k_pool(const float* __restrict__ fv, const float* __restrict__ fw,
                       float* __restrict__ pooled) {
  int i = blockIdx.x * blockDim.x + threadIdx.x;
  if (i >= B * FT) return;
  int b = i / FT, f = i % FT;
  float s = 0.f;
  if (f < FV) {
    for (int k = 0; k < KV; ++k) s += fv[((size_t)b * KV + k) * FV + f];
    s *= (1.0f / KV);
  } else {
    int g = f - FV;
    for (int k = 0; k < KW; ++k) s += fw[((size_t)b * KW + k) * FW + g];
    s *= (1.0f / KW);
  }
  pooled[i] = s;
}

// scores[b,k] = sum_h tanh(q[b,h] + kproj[b,k,h]) * wo[h] + bo
__global__ __launch_bounds__(128)
void k_attn_scores(const float* __restrict__ qproj, const float* __restrict__ kproj,
                   const float* __restrict__ wo, const float* __restrict__ bo,
                   float* __restrict__ scores, int Kc)
{
  int b = blockIdx.x, k = blockIdx.y, tid = threadIdx.x;
  const float* q  = qproj + (size_t)b * AH;
  const float* kp = kproj + ((size_t)b * Kc + k) * AH;
  float s = 0.f;
  for (int h = tid; h < AH; h += 128) s += tanhf(q[h] + kp[h]) * wo[h];
  __shared__ float red[128];
  red[tid] = s; __syncthreads();
  for (int o = 64; o; o >>= 1) { if (tid < o) red[tid] += red[tid + o]; __syncthreads(); }
  if (tid == 0) scores[b * Kc + k] = red[0] + bo[0];
}

// softmax over rows of length Kc (<=32); write to out[(b*T+t)*Kc + k]
__global__ __launch_bounds__(32)
void k_softmax_rows(const float* __restrict__ scores, float* __restrict__ outbase,
                    int Kc, int t)
{
  int b = blockIdx.x, k = threadIdx.x;
  __shared__ float sv[32];
  __shared__ float sm, ss;
  float v = (k < Kc) ? scores[b * Kc + k] : -3.0e38f;
  sv[k] = v;
  __syncthreads();
  if (k == 0) {
    float m = sv[0];
    for (int i = 1; i < Kc; ++i) m = fmaxf(m, sv[i]);
    float s = 0.f;
    for (int i = 0; i < Kc; ++i) s += expf(sv[i] - m);
    sm = m; ss = s;
  }
  __syncthreads();
  if (k < Kc) outbase[((size_t)b * T + t) * Kc + k] = expf(v - sm) / ss;
}

// g[b,f] = sigmoid(g[b,f]) * attended[b,f]
__global__ void k_attend_gate(const float* __restrict__ avs, const float* __restrict__ aws,
                              const float* __restrict__ fv, const float* __restrict__ fw,
                              float* __restrict__ g, int t)
{
  int i = blockIdx.x * blockDim.x + threadIdx.x;
  if (i >= B * FT) return;
  int b = i / FT, f = i % FT;
  float att = 0.f;
  if (f < FV) {
    const float* a = avs + ((size_t)b * T + t) * KV;
    for (int k = 0; k < KV; ++k) att += a[k] * fv[((size_t)b * KV + k) * FV + f];
  } else {
    int fx = f - FV;
    const float* a = aws + ((size_t)b * T + t) * KW;
    for (int k = 0; k < KW; ++k) att += a[k] * fw[((size_t)b * KW + k) * FW + fx];
  }
  g[i] = sigm(g[i]) * att;
}

// x[b,:] = concat(emb[prev[b],:], gated[b,:])
__global__ void k_build_x(const float* __restrict__ emb, const int* __restrict__ prev,
                          const float* __restrict__ gated, float* __restrict__ x)
{
  int i = blockIdx.x * blockDim.x + threadIdx.x;
  if (i >= B * XD) return;
  int b = i / XD, j = i % XD;
  x[i] = (j < E) ? emb[(size_t)prev[b] * E + j]
                 : gated[(size_t)b * FT + (j - E)];
}

// LSTM cell: gates layout [i|f|g|o] of width H each
__global__ void k_lstm(const float* __restrict__ gates, const float* __restrict__ c,
                       float* __restrict__ h2, float* __restrict__ c2)
{
  int i = blockIdx.x * blockDim.x + threadIdx.x;
  if (i >= B * H) return;
  int b = i / H, j = i % H;
  const float* g = gates + (size_t)b * G4;
  float ig = sigm(g[j]);
  float fg = sigm(g[H + j]);
  float gg = tanhf(g[2 * H + j]);
  float og = sigm(g[3 * H + j]);
  float cc = fg * c[i] + ig * gg;
  c2[i] = cc;
  h2[i] = og * tanhf(cc);
}

// log_softmax + copy mechanism + argmax, one block per batch row
__global__ __launch_bounds__(256)
void k_out_step(const float* __restrict__ logits, const float* __restrict__ h2,
                const float* __restrict__ wcopy, const float* __restrict__ bcopy,
                const int* __restrict__ annos, const float* __restrict__ aws,
                float* __restrict__ out_logp, float* __restrict__ out_tok,
                int* __restrict__ prev, int t)
{
  const int b = blockIdx.x, tid = threadIdx.x;
  __shared__ float red[256];
  __shared__ int   redi[256];
  __shared__ float s_logZ, s_pc;
  const float* L = logits + (size_t)b * V;

  // row max
  float m = -3.0e38f;
  for (int v = tid; v < V; v += 256) m = fmaxf(m, L[v]);
  red[tid] = m; __syncthreads();
  for (int o = 128; o; o >>= 1) { if (tid < o) red[tid] = fmaxf(red[tid], red[tid + o]); __syncthreads(); }
  float mx = red[0]; __syncthreads();

  // sum exp
  float s = 0.f;
  for (int v = tid; v < V; v += 256) s += expf(L[v] - mx);
  red[tid] = s; __syncthreads();
  for (int o = 128; o; o >>= 1) { if (tid < o) red[tid] += red[tid + o]; __syncthreads(); }
  if (tid == 0) s_logZ = mx + logf(red[0]);
  __syncthreads();

  // pc = sigmoid(h2 . wcopy + bcopy)
  float d = 0.f;
  for (int h = tid; h < H; h += 256) d += h2[(size_t)b * H + h] * wcopy[h];
  red[tid] = d; __syncthreads();
  for (int o = 128; o; o >>= 1) { if (tid < o) red[tid] += red[tid + o]; __syncthreads(); }
  if (tid == 0) s_pc = sigm(red[0] + bcopy[0]);
  __syncthreads();
  const float pc = s_pc, logZ = s_logZ;

  // probs = (1-pc)*softmax, scattered copy-attention added on top
  float* P = out_logp + ((size_t)b * T + t) * V;
  for (int v = tid; v < V; v += 256) P[v] = (1.0f - pc) * expf(L[v] - logZ);
  __threadfence_block(); __syncthreads();
  if (tid < KW) {
    int a = annos[b * KW + tid];
    atomicAdd(&P[a], pc * aws[((size_t)b * T + t) * KW + tid]);
  }
  __threadfence_block(); __syncthreads();

  // logp + argmax (first-max tie-break)
  float bm = -3.0e38f; int bi = 0;
  for (int v = tid; v < V; v += 256) {
    float lp = logf(P[v]);
    P[v] = lp;
    if (lp > bm) { bm = lp; bi = v; }
  }
  red[tid] = bm; redi[tid] = bi; __syncthreads();
  for (int o = 128; o; o >>= 1) {
    if (tid < o) {
      if (red[tid + o] > red[tid] ||
          (red[tid + o] == red[tid] && redi[tid + o] < redi[tid])) {
        red[tid] = red[tid + o]; redi[tid] = redi[tid + o];
      }
    }
    __syncthreads();
  }
  if (tid == 0) { out_tok[b * T + t] = (float)redi[0]; prev[b] = redi[0]; }
}

// ---------------------------------------------------------------------------
// Host orchestration
// ---------------------------------------------------------------------------
extern "C" void kernel_launch(void* const* d_in, const int* in_sizes, int n_in,
                              void* d_out, int out_size, void* d_ws, size_t ws_size,
                              hipStream_t stream)
{
  (void)in_sizes; (void)n_in; (void)out_size; (void)ws_size;
  const float* fv      = (const float*)d_in[0];
  const float* fw      = (const float*)d_in[1];
  const int*   annos   = (const int*)  d_in[2];
  /* d_in[3] = length, fixed at 15 by setup */
  const float* emb     = (const float*)d_in[4];
  const float* Wq_v    = (const float*)d_in[5];
  const float* bq_v    = (const float*)d_in[6];
  const float* Wk_v    = (const float*)d_in[7];
  const float* bk_v    = (const float*)d_in[8];
  const float* wo_v    = (const float*)d_in[9];
  const float* bo_v    = (const float*)d_in[10];
  const float* Wq_w    = (const float*)d_in[11];
  const float* bq_w    = (const float*)d_in[12];
  const float* Wk_w    = (const float*)d_in[13];
  const float* bk_w    = (const float*)d_in[14];
  const float* wo_w    = (const float*)d_in[15];
  const float* bo_w    = (const float*)d_in[16];
  const float* Wgate   = (const float*)d_in[17];
  const float* bgate   = (const float*)d_in[18];
  const float* Winit_h = (const float*)d_in[19];
  const float* binit_h = (const float*)d_in[20];
  const float* Winit_c = (const float*)d_in[21];
  const float* binit_c = (const float*)d_in[22];
  const float* wcopy   = (const float*)d_in[23];
  const float* bcopy   = (const float*)d_in[24];
  const float* W_ih    = (const float*)d_in[25];
  const float* b_ih    = (const float*)d_in[26];
  const float* W_hh    = (const float*)d_in[27];
  const float* b_hh    = (const float*)d_in[28];
  const float* Wout    = (const float*)d_in[29];
  const float* bout    = (const float*)d_in[30];

  float* out = (float*)d_out;
  float* ws  = (float*)d_ws;

  // ---- fp32 workspace ----
  size_t o = 0;
  float* pooled = ws + o; o += (size_t)B * FT;
  float* hA     = ws + o; o += (size_t)B * H;
  float* cA     = ws + o; o += (size_t)B * H;
  float* hB     = ws + o; o += (size_t)B * H;
  float* cB     = ws + o; o += (size_t)B * H;
  float* kprojv = ws + o; o += (size_t)B * KV * AH;
  float* kprojw = ws + o; o += (size_t)B * KW * AH;
  float* qv     = ws + o; o += (size_t)B * AH;
  float* qw     = ws + o; o += (size_t)B * AH;
  float* scor   = ws + o; o += (size_t)B * KW;
  float* gated  = ws + o; o += (size_t)B * FT;
  float* xbuf   = ws + o; o += (size_t)B * XD;
  float* gates  = ws + o; o += (size_t)B * G4;
  float* logits = ws + o; o += (size_t)B * V;
  int*   prev   = (int*)(ws + o); o += B;

  // ---- bf16 workspace (16B aligned; all rows are 64B multiples) ----
  o = (o + 3) & ~(size_t)3;
  unsigned short* us = (unsigned short*)(ws + o);
  size_t u = 0;
  unsigned short* bWq_v    = us + u; u += (size_t)512  * 512;
  unsigned short* bWq_w    = us + u; u += (size_t)512  * 512;
  unsigned short* bWk_v    = us + u; u += (size_t)512  * 2048;
  unsigned short* bWk_w    = us + u; u += (size_t)512  * FWp;
  unsigned short* bWgate   = us + u; u += (size_t)FTnp * 512;
  unsigned short* bWinit_h = us + u; u += (size_t)512  * FTp;
  unsigned short* bWinit_c = us + u; u += (size_t)512  * FTp;
  unsigned short* bW_ih    = us + u; u += (size_t)G4   * XDp;
  unsigned short* bW_hh    = us + u; u += (size_t)G4   * 512;
  unsigned short* bWout    = us + u; u += (size_t)Vnp  * 512;
  unsigned short* bpooled  = us + u; u += (size_t)B * FTp;
  unsigned short* bfv      = us + u; u += (size_t)B * KV * FV;
  unsigned short* bfw      = us + u; u += (size_t)B * KW * FWp;
  unsigned short* bh       = us + u; u += (size_t)B * H;
  unsigned short* bx       = us + u; u += (size_t)B * XDp;
  unsigned short* bh2      = us + u; u += (size_t)B * H;

  // ---- output layout (floats): logps | toks | avs | aws, all (B, T, ...) ----
  float* out_logp = out;
  float* out_tok  = out + (size_t)B * T * V;
  float* out_av   = out_tok + (size_t)B * T;
  float* out_aw   = out_av + (size_t)B * T * KV;

  auto conv = [&](const float* src, unsigned short* dst, int R, int K, int Rp, int Kp) {
    int tot = Rp * Kp;
    k_f32_to_bf16_pad<<<dim3((tot + 255) / 256), dim3(256), 0, stream>>>(src, dst, R, K, Kp, tot);
  };
  auto gemm = [&](const unsigned short* Ap, const unsigned short* Wp,
                  const float* bp, float* Cp, int M, int N, int Kp, int acc) {
    dim3 g((N + 31) / 32, M / 16);
    gemm_bf16_wmma<<<g, dim3(32), 0, stream>>>(Ap, Wp, bp, Cp, M, N, Kp, acc);
  };
  auto ew = [](int n) { return dim3((unsigned)((n + 255) / 256)); };

  // ---- one-time per call: weight + feature bf16 conversion ----
  k_init_prev<<<dim3(1), dim3(B), 0, stream>>>(prev);
  conv(Wq_v,    bWq_v,    512,  512,  512,  512);
  conv(Wq_w,    bWq_w,    512,  512,  512,  512);
  conv(Wk_v,    bWk_v,    512,  FV,   512,  FV);
  conv(Wk_w,    bWk_w,    512,  FW,   512,  FWp);
  conv(Wgate,   bWgate,   FT,   512,  FTnp, 512);
  conv(Winit_h, bWinit_h, 512,  FT,   512,  FTp);
  conv(Winit_c, bWinit_c, 512,  FT,   512,  FTp);
  conv(W_ih,    bW_ih,    G4,   XD,   G4,   XDp);
  conv(W_hh,    bW_hh,    G4,   512,  G4,   512);
  conv(Wout,    bWout,    V,    512,  Vnp,  512);
  conv(fv,      bfv,      B*KV, FV,   B*KV, FV);
  conv(fw,      bfw,      B*KW, FW,   B*KW, FWp);

  k_pool<<<ew(B * FT), dim3(256), 0, stream>>>(fv, fw, pooled);
  conv(pooled, bpooled, B, FT, B, FTp);
  gemm(bpooled, bWinit_h, binit_h, hA, B, H, FTp, 0);
  k_tanh_inplace<<<ew(B * H), dim3(256), 0, stream>>>(hA, B * H);
  gemm(bpooled, bWinit_c, binit_c, cA, B, H, FTp, 0);
  k_tanh_inplace<<<ew(B * H), dim3(256), 0, stream>>>(cA, B * H);
  gemm(bfv, bWk_v, bk_v, kprojv, B * KV, AH, FV, 0);   // key projections hoisted
  gemm(bfw, bWk_w, bk_w, kprojw, B * KW, AH, FWp, 0);  // out of the time loop

  float* h  = hA; float* c  = cA;
  float* h2 = hB; float* c2 = cB;

  // ---- 15 decode steps (serial through h/c) ----
  for (int t = 0; t < T; ++t) {
    conv(h, bh, B, H, B, H);
    gemm(bh, bWq_v, bq_v, qv, B, AH, H, 0);
    gemm(bh, bWq_w, bq_w, qw, B, AH, H, 0);
    k_attn_scores<<<dim3(B, KV), dim3(128), 0, stream>>>(qv, kprojv, wo_v, bo_v, scor, KV);
    k_softmax_rows<<<dim3(B), dim3(32), 0, stream>>>(scor, out_av, KV, t);
    k_attn_scores<<<dim3(B, KW), dim3(128), 0, stream>>>(qw, kprojw, wo_w, bo_w, scor, KW);
    k_softmax_rows<<<dim3(B), dim3(32), 0, stream>>>(scor, out_aw, KW, t);

    gemm(bh, bWgate, bgate, gated, B, FT, H, 0);
    k_attend_gate<<<ew(B * FT), dim3(256), 0, stream>>>(out_av, out_aw, fv, fw, gated, t);
    k_build_x<<<ew(B * XD), dim3(256), 0, stream>>>(emb, prev, gated, xbuf);
    conv(xbuf, bx, B, XD, B, XDp);

    gemm(bx, bW_ih, b_ih, gates, B, G4, XDp, 0);
    gemm(bh, bW_hh, b_hh, gates, B, G4, H, 1);   // accumulate
    k_lstm<<<ew(B * H), dim3(256), 0, stream>>>(gates, c, h2, c2);
    conv(h2, bh2, B, H, B, H);

    gemm(bh2, bWout, bout, logits, B, V, H, 0);
    k_out_step<<<dim3(B), dim3(256), 0, stream>>>(logits, h2, wcopy, bcopy, annos,
                                                  out_aw, out_logp, out_tok, prev, t);
    float* th = h; h = h2; h2 = th;
    float* tc = c; c = c2; c2 = tc;
  }
}
